// GraphSAGEModel_45655502356568
// MI455X (gfx1250) — matrix operand order repaired and checked by reading, not compiled, not run on hardware
//
#include <hip/hip_runtime.h>
#include <hip/hip_bf16.h>

// GraphSAGE 2-layer forward for MI455X (gfx1250, wave32).
//
// Roofline: edge scatter-mean moves ~3.3 GB (L2-resident: x/h are 51.2 MB
// each, well inside the 192 MB L2) -> memory-bound part. GEMMs are only
// 13 GFLOP total -> v_wmma_f32_16x16x32_bf16 makes them free next to the
// scatter traffic. Fusions: mean-divide into A-fragment load, bias +
// L2-normalize + ReLU into the WMMA epilogue. A-fragment f32->bf16 uses
// hardware converts (__bf16 casts) instead of integer bit-twiddling.

#define N_NODES 100000
#define N_EDGES 1600000
#define DIM     128

typedef __attribute__((ext_vector_type(16))) __bf16        v16bf;
typedef __attribute__((ext_vector_type(8)))  float         v8f;
typedef __attribute__((ext_vector_type(8)))  unsigned int  v8u;

__device__ __forceinline__ unsigned short f2bf_bits(float f) {
    // round-to-nearest-even f32 -> bf16 (used only in tiny weight-prep kernel)
    unsigned u = __float_as_uint(f);
    unsigned r = u + 0x7FFFu + ((u >> 16) & 1u);
    return (unsigned short)(r >> 16);
}

// ---------------------------------------------------------------------------
// Scatter-add: one wave per edge; 32 lanes x float4 = 128 dims.
// agg[dst] += x[src]; cnt[dst] += 1
// ---------------------------------------------------------------------------
__global__ void sage_scatter(const float* __restrict__ X,
                             const int*   __restrict__ src,
                             const int*   __restrict__ dst,
                             float*       __restrict__ agg,
                             unsigned int* __restrict__ cnt)
{
    int t    = blockIdx.x * blockDim.x + threadIdx.x;
    int e    = t >> 5;
    int lane = t & 31;
    if (e >= N_EDGES) return;                 // uniform per wave
    int s = src[e];
    int d = dst[e];
    float4 v = ((const float4*)(X + (size_t)s * DIM))[lane];
    float* ar = agg + (size_t)d * DIM + lane * 4;
    unsafeAtomicAdd(ar + 0, v.x);             // global_atomic_add_f32
    unsafeAtomicAdd(ar + 1, v.y);
    unsafeAtomicAdd(ar + 2, v.z);
    unsafeAtomicAdd(ar + 3, v.w);
    if (lane == 0) atomicAdd(cnt + d, 1u);
}

// inv[i] = 1 / max(cnt[i], 1)   (fused into GEMM A-load as row scale)
__global__ void sage_invcnt(const unsigned int* __restrict__ cnt,
                            float* __restrict__ inv)
{
    int i = blockIdx.x * blockDim.x + threadIdx.x;
    if (i < N_NODES) inv[i] = 1.0f / fmaxf((float)cnt[i], 1.0f);
}

// Pre-transpose + bf16-convert weights: Wt[n][k] for k in [0,256)
// (k<128 -> Wl[k][n], else Wr[k-128][n]).  64 KB, stays hot in L2/WGP$.
__global__ void sage_wt(const float* __restrict__ Wl,
                        const float* __restrict__ Wr,
                        unsigned short* __restrict__ Wt)
{
    int idx = blockIdx.x * blockDim.x + threadIdx.x;   // 0..32767
    int n = idx >> 8;
    int k = idx & 255;
    float v = (k < 128) ? Wl[(size_t)k * DIM + n] : Wr[(size_t)(k - 128) * DIM + n];
    Wt[(size_t)n * 256 + k] = f2bf_bits(v);
}

// ---------------------------------------------------------------------------
// Fused GEMM: out = normalize( (agg*inv) @ Wl + X @ Wr + b ), optional ReLU.
// Block = 256 threads = 8 waves; each wave -> 16 rows x 128 cols
// (8 v8f accumulators). K = 256 in 8 steps of 32 -> 64 WMMA/wave.
// ---------------------------------------------------------------------------
__global__ void __launch_bounds__(256)
sage_gemm(const float* __restrict__ agg,
          const float* __restrict__ inv,
          const float* __restrict__ X,
          const unsigned short* __restrict__ Wt,
          const float* __restrict__ bias,
          float* __restrict__ out,
          int relu)
{
    const int lane = threadIdx.x & 31;
    const int wave = threadIdx.x >> 5;
    const int half = lane >> 4;      // 0: lanes 0-15, 1: lanes 16-31
    const int l16  = lane & 15;

    // A-fragment row (ISA 16-bit A layout: both lane halves cover rows 0-15,
    // differing in K range)
    int rowA  = blockIdx.x * 128 + wave * 16 + l16;
    int rowAc = min(rowA, N_NODES - 1);
    const float aScale = inv[rowAc];

    v8f acc[8];
#pragma unroll
    for (int n = 0; n < 8; ++n)
        acc[n] = (v8f){0.f, 0.f, 0.f, 0.f, 0.f, 0.f, 0.f, 0.f};

#pragma unroll
    for (int step = 0; step < 8; ++step) {
        const int kk = step * 32;
        const float* srcRow;
        float sc;
        int klocal;
        if (kk < 128) { srcRow = agg + (size_t)rowAc * DIM; sc = aScale; klocal = kk; }
        else          { srcRow = X   + (size_t)rowAc * DIM; sc = 1.0f;   klocal = kk - 128; }

        // A fragment: element e -> K = kk + half*16 + e
        const int k0 = klocal + half * 16;
        float4 a0 = *(const float4*)(srcRow + k0);
        float4 a1 = *(const float4*)(srcRow + k0 + 4);
        float4 a2 = *(const float4*)(srcRow + k0 + 8);
        float4 a3 = *(const float4*)(srcRow + k0 + 12);
        // Hardware f32->bf16 converts (v_cvt_*_bf16) instead of bit ops
        v16bf Av;
        Av[0]  = (__bf16)(a0.x * sc); Av[1]  = (__bf16)(a0.y * sc);
        Av[2]  = (__bf16)(a0.z * sc); Av[3]  = (__bf16)(a0.w * sc);
        Av[4]  = (__bf16)(a1.x * sc); Av[5]  = (__bf16)(a1.y * sc);
        Av[6]  = (__bf16)(a1.z * sc); Av[7]  = (__bf16)(a1.w * sc);
        Av[8]  = (__bf16)(a2.x * sc); Av[9]  = (__bf16)(a2.y * sc);
        Av[10] = (__bf16)(a2.z * sc); Av[11] = (__bf16)(a2.w * sc);
        Av[12] = (__bf16)(a3.x * sc); Av[13] = (__bf16)(a3.y * sc);
        Av[14] = (__bf16)(a3.z * sc); Av[15] = (__bf16)(a3.w * sc);

#pragma unroll
        for (int n = 0; n < 8; ++n) {
            // B fragment (same K order as A): lane -> col, 16 consecutive K
            // = one contiguous 32-byte load from transposed bf16 weights.
            const int col = n * 16 + l16;
            union { v8u q; v16bf v; } B;
            B.q = *(const v8u*)(Wt + (size_t)col * 256 + kk + half * 16);
            acc[n] = __builtin_amdgcn_wmma_f32_16x16x32_bf16(
                false, Av, false, B.v, (short)0, acc[n], false, false);
        }
    }

    // Epilogue: bias, row L2-norm (shfl within 16-lane halves), ReLU, store.
    float ss[8];
#pragma unroll
    for (int r = 0; r < 8; ++r) ss[r] = 0.f;
#pragma unroll
    for (int n = 0; n < 8; ++n) {
        float bv = bias[n * 16 + l16];
#pragma unroll
        for (int r = 0; r < 8; ++r) {
            float v = acc[n][r] + bv;
            acc[n][r] = v;
            ss[r] += v * v;
        }
    }
#pragma unroll
    for (int r = 0; r < 8; ++r) {
        float s = ss[r];
        s += __shfl_xor(s, 1);
        s += __shfl_xor(s, 2);
        s += __shfl_xor(s, 4);
        s += __shfl_xor(s, 8);              // stays within each 16-lane half
        ss[r] = 1.0f / fmaxf(sqrtf(s), 1e-12f);
    }
#pragma unroll
    for (int n = 0; n < 8; ++n) {
        const int col = n * 16 + l16;
#pragma unroll
        for (int r = 0; r < 8; ++r) {
            int row = blockIdx.x * 128 + wave * 16 + r + half * 8;  // C/D layout
            if (row < N_NODES) {
                float v = acc[n][r] * ss[r];
                if (relu) v = fmaxf(v, 0.f);
                out[(size_t)row * DIM + col] = v;
            }
        }
    }
}

// ---------------------------------------------------------------------------
extern "C" void kernel_launch(void* const* d_in, const int* in_sizes, int n_in,
                              void* d_out, int out_size, void* d_ws, size_t ws_size,
                              hipStream_t stream)
{
    const float* x   = (const float*)d_in[0];
    const int*   ei  = (const int*)d_in[1];     // [2][E] int32
    const float* W1l = (const float*)d_in[2];
    const float* b1  = (const float*)d_in[3];
    const float* W1r = (const float*)d_in[4];
    const float* W2l = (const float*)d_in[5];
    const float* b2  = (const float*)d_in[6];
    const float* W2r = (const float*)d_in[7];
    float*       out = (float*)d_out;

    const int* src = ei;
    const int* dst = ei + N_EDGES;

    // Workspace layout (d_ws is 256B aligned; ~103 MB used)
    char* ws = (char*)d_ws;
    const size_t featB = (size_t)N_NODES * DIM * 4;          // 51.2 MB
    float*          agg = (float*)ws;                         // [N,128] f32
    unsigned int*   cnt = (unsigned int*)(ws + featB);        // [N] u32
    float*          inv = (float*)(ws + featB + (size_t)N_NODES * 4);
    float*          h   = (float*)(ws + featB + (size_t)N_NODES * 8);
    unsigned short* Wt  = (unsigned short*)(ws + 2 * featB + (size_t)N_NODES * 8);

    const int scatterBlocks = (N_EDGES * 32 + 255) / 256;     // 1 wave / edge
    const int nodeBlocks    = (N_NODES + 255) / 256;
    const int gemmBlocks    = (N_NODES + 127) / 128;

    // ---- layer 1 ----
    hipMemsetAsync(agg, 0, featB + (size_t)N_NODES * 4, stream);  // agg+cnt
    sage_scatter<<<scatterBlocks, 256, 0, stream>>>(x, src, dst, agg, cnt);
    sage_invcnt <<<nodeBlocks,    256, 0, stream>>>(cnt, inv);
    sage_wt     <<<128,           256, 0, stream>>>(W1l, W1r, Wt);
    sage_gemm   <<<gemmBlocks,    256, 0, stream>>>(agg, inv, x, Wt, b1, h, 1);

    // ---- layer 2 ----
    hipMemsetAsync(agg, 0, featB + (size_t)N_NODES * 4, stream);
    sage_scatter<<<scatterBlocks, 256, 0, stream>>>(h, src, dst, agg, cnt);
    sage_invcnt <<<nodeBlocks,    256, 0, stream>>>(cnt, inv);
    sage_wt     <<<128,           256, 0, stream>>>(W2l, W2r, Wt);
    sage_gemm   <<<gemmBlocks,    256, 0, stream>>>(agg, inv, h, Wt, b2, out, 0);
}
